// Mixture_24541443129646
// MI455X (gfx1250) — matrix operand order repaired and checked by reading, not compiled, not run on hardware
//
#include <hip/hip_runtime.h>

// Problem constants (from reference): N=131072 rows, K=256 components, D=128 dims
#define DIM   128
#define NCOMP 256
#define NROWS 131072
#define NSTRIPS (NROWS / 16)      // 8192 strips of 16 rows, one wave each
#define WAVES_PER_BLOCK 16        // 512 threads; block stages B once for 16 waves
#define NBLOCKS (NSTRIPS / WAVES_PER_BLOCK)    // 512

typedef __attribute__((ext_vector_type(16))) __bf16 v16bf;
typedef __attribute__((ext_vector_type(8)))  float  v8f;
typedef __attribute__((ext_vector_type(4)))  float  v4f;

// Packed B operand store: [hl(2)][chunk(4)][ktile(16)][sub(2)][lane(32)] x 16B
// (sub = which 16B half of the 32B per-lane operand; lane stride = 16B so the
//  ds_load_b128 reads are conflict-free across the 64 LDS banks)
#define B_ENTRIES (2 * 4 * 16 * 2 * 32)     // 8192 int4 entries = 128 KB
#define B_LO_ENT  (4 * 16 * 2 * 32)         // entry offset of the "lo" half

// Workspace layout (~161 KB):
#define SS_OFF    0                  // float strip_sums[8192]   32 KB
#define MUSQ_OFF  (32 * 1024)        // float musq[256]           1 KB
#define B_OFF     (33 * 1024)        // int4  bpack[8192]       128 KB

#define SMEM_BYTES (128 * 1024 + 512)   // B in LDS + prec[128]

// ---------------------------------------------------------------------------
// Pass 0: fold prec into mu, split into bf16 hi/lo, pre-pack into the WMMA
// B-operand layout for v_wmma_f32_16x16x32_bf16 (B is 32x16, per-lane:
// col = lane&15, kbase = 8*(lane>>4), element e <-> K = kbase+e (e<8), 8+kbase+e (e>=8)).
// Also musq[k] = sum_d prec[d]*mu[k,d]^2 in exact f32.
// ---------------------------------------------------------------------------
__global__ void prep_kernel(const float* __restrict__ mu,
                            const float* __restrict__ prec,
                            float* __restrict__ musq,
                            __bf16* __restrict__ bpack) {
    int tid = blockIdx.x * blockDim.x + threadIdx.x;
    if (tid < 4 * 16 * 32) {               // (chunk, ktile, lane)
        int c = tid >> 9;
        int t = (tid >> 5) & 15;
        int l = tid & 31;
        int col   = t * 16 + (l & 15);
        int kbase = (l >> 4) * 8;
        #pragma unroll
        for (int e = 0; e < 16; ++e) {
            int d = c * 32 + kbase + (e < 8 ? e : e + 8);
            float v = prec[d] * mu[col * DIM + d];
            __bf16 h  = (__bf16)v;
            __bf16 lo = (__bf16)(v - (float)h);
            int sub = e >> 3, pos = e & 7;
            int ehi = ((c * 16 + t) * 2 + sub) * 32 + l;               // hl=0
            int elo = ehi + B_LO_ENT;                                  // hl=1
            bpack[ehi * 8 + pos] = h;
            bpack[elo * 8 + pos] = lo;
        }
    }
    if (tid < NCOMP) {
        float s = 0.f;
        for (int d = 0; d < DIM; ++d) {
            float m = mu[tid * DIM + d];
            s = fmaf(prec[d] * m, m, s);
        }
        musq[tid] = s;
    }
}

// ---------------------------------------------------------------------------
// Main: block (16 waves) stages all B operands (128 KB) into LDS once, then
// each wave32 processes ONE strip of 16 rows (16 accumulator tiles of 16x16
// f32 cover all K=256) -- no per-wave loop, so no loop-invariant LDS loads
// for LICM to hoist/spill. 3x-bf16-split GEMM over D in 4 chunks of 32, then
// in-register logsumexp over K per row. x is read exactly once.
// ---------------------------------------------------------------------------
__global__ __launch_bounds__(512) void gmm_lse_kernel(
        const float* __restrict__ x,
        const float* __restrict__ prec,
        const float* __restrict__ musq,
        const int4* __restrict__ gB,
        float* __restrict__ strip_sums) {
    extern __shared__ char smem[];
    int4*  sB     = (int4*)smem;
    float* s_prec = (float*)(smem + 128 * 1024);

    // Stage packed B (hi+lo) into LDS, coalesced b128 copies.
    for (int i = threadIdx.x; i < B_ENTRIES; i += 512) sB[i] = gB[i];
    if (threadIdx.x < DIM) s_prec[threadIdx.x] = prec[threadIdx.x];
    __syncthreads();

    int lane  = threadIdx.x & 31;
    int wave  = threadIdx.x >> 5;
    int strip = blockIdx.x * WAVES_PER_BLOCK + wave;
    int kb    = (lane < 16) ? 0 : 8;    // A operand K-run base for this lane
    int half8 = (lane >> 4) * 8;

    int row = strip * 16 + (lane & 15);
    const float* xr = x + (size_t)row * DIM;

    // Per-lane musq values (k = tile*16 + lane&15)
    float mq[16];
    #pragma unroll
    for (int t = 0; t < 16; ++t) mq[t] = musq[t * 16 + (lane & 15)];

    union BU { int4 q[2]; v16bf v; };
    v8f zero = {0.f, 0.f, 0.f, 0.f, 0.f, 0.f, 0.f, 0.f};
    v8f acc[16];
    #pragma unroll
    for (int t = 0; t < 16; ++t) acc[t] = zero;

    float xsq = 0.f;

    #pragma unroll
    for (int c = 0; c < 4; ++c) {
        int d0 = c * 32 + kb;
        v4f a0 = *(const v4f*)(xr + d0);
        v4f a1 = *(const v4f*)(xr + d0 + 4);
        v4f a2 = *(const v4f*)(xr + d0 + 16);
        v4f a3 = *(const v4f*)(xr + d0 + 20);
        float av[16];
        #pragma unroll
        for (int j = 0; j < 4; ++j) {
            av[j] = a0[j]; av[4 + j] = a1[j];
            av[8 + j] = a2[j]; av[12 + j] = a3[j];
        }
        v16bf ahi, alo;
        #pragma unroll
        for (int e = 0; e < 16; ++e) {
            int d = d0 + (e < 8 ? e : e + 8);
            float v = av[e];
            xsq = fmaf(s_prec[d] * v, v, xsq);
            __bf16 h = (__bf16)v;
            ahi[e] = h;
            alo[e] = (__bf16)(v - (float)h);
        }
        #pragma unroll
        for (int t = 0; t < 16; ++t) {
            int hbase = ((c * 16 + t) * 2) * 32 + lane;
            BU bh, bl;
            bh.q[0] = sB[hbase];
            bh.q[1] = sB[hbase + 32];
            bl.q[0] = sB[hbase + B_LO_ENT];
            bl.q[1] = sB[hbase + B_LO_ENT + 32];
            // cross ~= hi*hi + hi*lo + lo*hi  (drop lo*lo, ~2^-17 rel)
            acc[t] = __builtin_amdgcn_wmma_f32_16x16x32_bf16(
                         false, ahi, false, bh.v, (short)0, acc[t], false, false);
            acc[t] = __builtin_amdgcn_wmma_f32_16x16x32_bf16(
                         false, ahi, false, bl.v, (short)0, acc[t], false, false);
            acc[t] = __builtin_amdgcn_wmma_f32_16x16x32_bf16(
                         false, alo, false, bh.v, (short)0, acc[t], false, false);
        }
    }

    // Complete x_sq: this lane + xor-16 partner cover the full row.
    xsq += __shfl_xor(xsq, 16, 32);
    // C tile layout: VGPR s, lane L -> (M = s + 8*(L>>4), N = L&15);
    // gather x_sq of the row each (s,lane) slot refers to.
    float xs_slot[8];
    #pragma unroll
    for (int s = 0; s < 8; ++s) xs_slot[s] = __shfl(xsq, half8 + s, 32);

    float rm[8];
    #pragma unroll
    for (int s = 0; s < 8; ++s) rm[s] = -3.4e38f;
    #pragma unroll
    for (int t = 0; t < 16; ++t) {
        #pragma unroll
        for (int s = 0; s < 8; ++s) {
            float m = acc[t][s] - 0.5f * (xs_slot[s] + mq[t]);
            rm[s] = fmaxf(rm[s], m);
        }
    }
    #pragma unroll
    for (int s = 0; s < 8; ++s) {
        rm[s] = fmaxf(rm[s], __shfl_xor(rm[s], 1, 32));
        rm[s] = fmaxf(rm[s], __shfl_xor(rm[s], 2, 32));
        rm[s] = fmaxf(rm[s], __shfl_xor(rm[s], 4, 32));
        rm[s] = fmaxf(rm[s], __shfl_xor(rm[s], 8, 32));
    }
    float se[8];
    #pragma unroll
    for (int s = 0; s < 8; ++s) se[s] = 0.f;
    #pragma unroll
    for (int t = 0; t < 16; ++t) {
        #pragma unroll
        for (int s = 0; s < 8; ++s) {
            float m = acc[t][s] - 0.5f * (xs_slot[s] + mq[t]);
            se[s] += __expf(m - rm[s]);
        }
    }
    #pragma unroll
    for (int s = 0; s < 8; ++s) {
        se[s] += __shfl_xor(se[s], 1, 32);
        se[s] += __shfl_xor(se[s], 2, 32);
        se[s] += __shfl_xor(se[s], 4, 32);
        se[s] += __shfl_xor(se[s], 8, 32);
    }
    float tot = 0.f;
    #pragma unroll
    for (int s = 0; s < 8; ++s) tot += rm[s] + __logf(se[s]);
    tot += __shfl_xor(tot, 16, 32);   // combine row halves 0-7 / 8-15
    if (lane == 0) strip_sums[strip] = tot;
}

// ---------------------------------------------------------------------------
// Deterministic final reduction (double accumulation), out = -sum(lse)
// ---------------------------------------------------------------------------
__global__ void finalize_kernel(const float* __restrict__ ss,
                                float* __restrict__ out, int n) {
    __shared__ double sh[256];
    double s = 0.0;
    for (int i = threadIdx.x; i < n; i += 256) s += (double)ss[i];
    sh[threadIdx.x] = s;
    __syncthreads();
    for (int off = 128; off > 0; off >>= 1) {
        if ((int)threadIdx.x < off) sh[threadIdx.x] += sh[threadIdx.x + off];
        __syncthreads();
    }
    if (threadIdx.x == 0) out[0] = (float)(-sh[0]);
}

extern "C" void kernel_launch(void* const* d_in, const int* in_sizes, int n_in,
                              void* d_out, int out_size, void* d_ws, size_t ws_size,
                              hipStream_t stream) {
    const float* x    = (const float*)d_in[0];   // (N, D) f32
    const float* mu   = (const float*)d_in[1];   // (K, D) f32
    const float* prec = (const float*)d_in[2];   // (D,)   f32

    char*   ws         = (char*)d_ws;
    float*  strip_sums = (float*)(ws + SS_OFF);
    float*  musq       = (float*)(ws + MUSQ_OFF);
    __bf16* bpack      = (__bf16*)(ws + B_OFF);

    prep_kernel<<<8, 256, 0, stream>>>(mu, prec, musq, bpack);
    gmm_lse_kernel<<<NBLOCKS, 512, SMEM_BYTES, stream>>>(
        x, prec, musq, (const int4*)(ws + B_OFF), strip_sums);
    finalize_kernel<<<1, 256, 0, stream>>>(strip_sums, (float*)d_out, NSTRIPS);
}
